// TFGridNet_M_43739946943056
// MI455X (gfx1250) — matrix operand order repaired and checked by reading, not compiled, not run on hardware
//
#include <hip/hip_runtime.h>
#include <hip/hip_bf16.h>

// ---------------------------------------------------------------------------
// TFGridNet block for MI455X (gfx1250, wave32).
// All large matmuls use v_wmma_f32_16x16x32_bf16 (f32 accumulate).
// LSTM recurrences run as persistent WMMA scan kernels (16 seqs / workgroup,
// fused [x|h]@[Wih|Whh]^T, K=384, weights streamed from L2 each step).
// ConvT1d is a sliding-window GEMM (K=1536).
// Workspace budget ~250 MB (bump allocated from d_ws).
// ---------------------------------------------------------------------------

typedef __bf16 bf16;
typedef __attribute__((ext_vector_type(16))) __bf16 bf16x16;
typedef __attribute__((ext_vector_type(8)))  float   f32x8;

static constexpr int Bd  = 2;
static constexpr int Cd  = 48;
static constexpr int Td  = 500;
static constexpr int Fd  = 65;
static constexpr int HLd = 192;   // hidden
static constexpr int G4  = 768;   // 4*HL
static constexpr int INd = 192;   // C*KS
static constexpr int LA  = 62;    // intra Lout (Fd-4+1)
static constexpr int LB  = 497;   // inter Lout (Td-4+1)
static constexpr int NSA = Bd * Td;   // 1000 intra sequences
static constexpr int NSB = Bd * Fd;   // 130 inter sequences
static constexpr int PTA = LA + 6;    // padded time rows (intra) = 68
static constexpr int PTB = LB + 6;    // padded time rows (inter) = 503
static constexpr int NHd = 4, Ed = 8, CVd = 12;
static constexpr int DQ  = Ed * Fd;   // 520
static constexpr int DV  = CVd * Fd;  // 780
static constexpr int MR  = Bd * Td * Fd; // 65000 "pixel" rows

__device__ __forceinline__ float sigm(float x){ return 1.0f / (1.0f + __expf(-x)); }

__device__ __forceinline__ float blockSum(float v, float* sh){
  int tid = threadIdx.x;
  sh[tid] = v; __syncthreads();
  for (int s = 128; s > 0; s >>= 1) { if (tid < s) sh[tid] += sh[tid + s]; __syncthreads(); }
  float r = sh[0]; __syncthreads();
  return r;
}
__device__ __forceinline__ float blockMax(float v, float* sh){
  int tid = threadIdx.x;
  sh[tid] = v; __syncthreads();
  for (int s = 128; s > 0; s >>= 1) { if (tid < s) sh[tid] = fmaxf(sh[tid], sh[tid + s]); __syncthreads(); }
  float r = sh[0]; __syncthreads();
  return r;
}

// ---------------- elementwise / prep kernels -------------------------------

// LayerNorm over channel dim of (B,C,T,F), per (b,t,f).
__global__ __launch_bounds__(256) void k_ln_channel(
    const float* __restrict__ x, const float* __restrict__ g,
    const float* __restrict__ b, float* __restrict__ y)
{
  int idx = blockIdx.x * 256 + threadIdx.x;
  if (idx >= Bd * Td * Fd) return;
  int f = idx % Fd, t = (idx / Fd) % Td, bb = idx / (Fd * Td);
  size_t base = ((size_t)bb * Cd) * Td * Fd + (size_t)t * Fd + f;
  const size_t cs = (size_t)Td * Fd;
  float mu = 0.f;
  for (int c = 0; c < Cd; ++c) mu += x[base + c * cs];
  mu *= (1.0f / Cd);
  float var = 0.f;
  for (int c = 0; c < Cd; ++c) { float d = x[base + c * cs] - mu; var += d * d; }
  var *= (1.0f / Cd);
  float rs = rsqrtf(var + 1e-5f);
  for (int c = 0; c < Cd; ++c)
    y[base + c * cs] = (x[base + c * cs] - mu) * rs * g[c] + b[c];
}

// intra unfold: (b,t) sequences over f.  out[(n,l,j)] with j=c*4+k.
__global__ __launch_bounds__(256) void k_unfold_intra(
    const float* __restrict__ u, bf16* __restrict__ o, int flip)
{
  long long idx = (long long)blockIdx.x * 256 + threadIdx.x;
  if (idx >= (long long)NSA * LA * INd) return;
  int j = (int)(idx % INd);
  int l = (int)((idx / INd) % LA);
  int n = (int)(idx / ((long long)INd * LA));
  int b = n / Td, t = n % Td;
  int c = j >> 2, k = j & 3;
  int le = flip ? (LA - 1 - l) : l;
  int fidx = le + k;
  o[idx] = (bf16)u[(((size_t)b * Cd + c) * Td + t) * Fd + fidx];
}

// inter unfold: (b,f) sequences over t.
__global__ __launch_bounds__(256) void k_unfold_inter(
    const float* __restrict__ u, bf16* __restrict__ o, int flip)
{
  long long idx = (long long)blockIdx.x * 256 + threadIdx.x;
  if (idx >= (long long)NSB * LB * INd) return;
  int j = (int)(idx % INd);
  int l = (int)((idx / INd) % LB);
  int n = (int)(idx / ((long long)INd * LB));
  int b = n / Fd, f = n % Fd;
  int c = j >> 2, k = j & 3;
  int le = flip ? (LB - 1 - l) : l;
  int tidx = le + k;
  o[idx] = (bf16)u[(((size_t)b * Cd + c) * Td + tidx) * Fd + f];
}

// combined LSTM weight, n-major: W[n*384+k] = k<192 ? Wih[n][k] : Whh[n][k-192]
__global__ __launch_bounds__(256) void k_combine_w(
    const float* __restrict__ Wih, const float* __restrict__ Whh, bf16* __restrict__ o)
{
  int idx = blockIdx.x * 256 + threadIdx.x;
  if (idx >= G4 * 2 * HLd) return;
  int n = idx / (2 * HLd), k = idx % (2 * HLd);
  float v = (k < HLd) ? Wih[(size_t)n * INd + k] : Whh[(size_t)n * HLd + (k - HLd)];
  o[idx] = (bf16)v;
}

// convT weight, n-major (48 x 1536): Wcat[n][j*384+i] = Wt[i][n][3-j]
__global__ __launch_bounds__(256) void k_wcat(
    const float* __restrict__ Wt, bf16* __restrict__ o)
{
  int idx = blockIdx.x * 256 + threadIdx.x;
  if (idx >= Cd * 1536) return;
  int n = idx / 1536, kk = idx % 1536;
  int j = kk / 384, i = kk % 384;
  o[idx] = (bf16)Wt[((size_t)i * Cd + n) * 4 + (3 - j)];
}

__global__ __launch_bounds__(256) void k_f2b(
    const float* __restrict__ s, bf16* __restrict__ d, long long n)
{
  long long i = (long long)blockIdx.x * 256 + threadIdx.x;
  if (i < n) d[i] = (bf16)s[i];
}

__global__ __launch_bounds__(256) void k_zero_bf16(bf16* __restrict__ p, long long n)
{
  long long i = (long long)blockIdx.x * 256 + threadIdx.x;
  if (i < n) p[i] = (bf16)0.0f;
}

// ---------------- LSTM persistent scan (WMMA) ------------------------------
// 192 threads = 6 waves; 16 sequences per workgroup; per step the block
// computes gates(16x768) = [x_t | h](16x384) @ W^T via 576 bf16 WMMAs.
// W is n-major (768 x 384), streamed from L2 every step (an opaque asm
// dependence stops LICM from hoisting+spilling the 96 weight fragments).
__global__ __launch_bounds__(192) void k_lstm_scan(
    const bf16* __restrict__ X,     // (Nseq, L, 192)
    const bf16* __restrict__ W,     // (768, 384) n-major
    const float* __restrict__ bias, // (768)
    bf16* __restrict__ Out,         // (Nseq, padT, 384)
    int Nseq, int L, int padT, int flip, int chanOff)
{
  __shared__ __align__(16) bf16 hA[16][384];
  __shared__ float cS[16][192];
  const int tid  = threadIdx.x;
  const int lane = tid & 31;
  const int wave = tid >> 5;              // 0..5
  const int seqBase = blockIdx.x * 16;

  for (int i = tid; i < 16 * 192; i += 192) {
    int s = i / 192, j = i % 192;
    hA[s][192 + j] = (bf16)0.0f;
    cS[s][j] = 0.0f;
  }
  __syncthreads();

  const int m     = lane & 15;
  const int half  = lane >> 4;
  const int abase = half * 8;
  const int kb    = half * 16;
  const int nlane = lane & 15;

  f32x8 zero8;
  #pragma unroll
  for (int e = 0; e < 8; ++e) zero8[e] = 0.0f;

  for (int t = 0; t < L; ++t) {
    // stage x_t
    for (int i = tid; i < 16 * 192; i += 192) {
      int s = i / 192, j = i % 192;
      int seq = seqBase + s; if (seq >= Nseq) seq = Nseq - 1;
      hA[s][j] = X[((size_t)seq * L + t) * 192 + j];
    }
    __syncthreads();

    // Opaque zero: makes the weight pointer loop-variant so the compiler
    // re-loads B fragments from global (L2) each timestep instead of
    // hoisting 96 fragments out of the loop and spilling them to scratch.
    unsigned wzero = 0;
    asm volatile("" : "+v"(wzero));
    const bf16* Wt = W + wzero;

    f32x8 acc[2][4];
    #pragma unroll
    for (int ht = 0; ht < 2; ++ht)
      #pragma unroll
      for (int g = 0; g < 4; ++g) acc[ht][g] = zero8;

    for (int kt = 0; kt < 12; ++kt) {
      int k0 = kt * 32;
      bf16x16 a;
      #pragma unroll
      for (int e = 0; e < 8; ++e) {
        a[e]     = hA[m][k0 + kb + e];
        a[e + 8] = hA[m][k0 + kb + 8 + e];
      }
      #pragma unroll
      for (int ht = 0; ht < 2; ++ht) {
        int hidTile = wave * 2 + ht;
        #pragma unroll
        for (int g = 0; g < 4; ++g) {
          int n0 = g * HLd + hidTile * 16;
          bf16x16 bfr = *(const bf16x16*)(&Wt[(size_t)(n0 + nlane) * 384 + k0 + kb]);
          acc[ht][g] = __builtin_amdgcn_wmma_f32_16x16x32_bf16(
              false, a, false, bfr, (short)0, acc[ht][g], false, false);
        }
      }
    }
    __syncthreads();   // all waves done reading hA

    #pragma unroll
    for (int ht = 0; ht < 2; ++ht) {
      int hidTile = wave * 2 + ht;
      #pragma unroll
      for (int r = 0; r < 8; ++r) {
        int row = r + half * 8;          // sequence slot
        int hid = hidTile * 16 + nlane;  // hidden index
        float iv = acc[ht][0][r] + bias[0 * HLd + hid];
        float fv = acc[ht][1][r] + bias[1 * HLd + hid];
        float gv = acc[ht][2][r] + bias[2 * HLd + hid];
        float ov = acc[ht][3][r] + bias[3 * HLd + hid];
        float cn = sigm(fv) * cS[row][hid] + sigm(iv) * tanhf(gv);
        float hn = sigm(ov) * tanhf(cn);
        cS[row][hid] = cn;
        bf16 hb = (bf16)hn;
        hA[row][192 + hid] = hb;
        int seq = seqBase + row;
        if (seq < Nseq) {
          int tout = flip ? (L - 1 - t) : t;
          Out[((size_t)seq * padT + 3 + tout) * 384 + chanOff + hid] = hb;
        }
      }
    }
    __syncthreads();
  }
}

// ---------------- generic batched WMMA GEMM --------------------------------
// C(M,N) f32 = A(M,K) bf16 . B(N,K n-major) bf16 + bias[N]
// aMode 1: A row r maps to a sliding window in a padded image:
//          off = ((r/rowsPerImg)*padT + r%rowsPerImg)*rowStride, length K.
__global__ __launch_bounds__(256) void k_gemm(
    int M, int N, int K,
    const bf16* __restrict__ A, long long aBatch,
    int aMode, int rowsPerImg, int padT, int rowStride,
    const bf16* __restrict__ B, long long bBatch,
    float* __restrict__ C, long long cBatch,
    const float* __restrict__ bias)
{
  int ntiles = (N + 15) >> 4;
  int mtiles = (M + 15) >> 4;
  int wave = threadIdx.x >> 5;
  int lane = threadIdx.x & 31;
  long long tile = (long long)blockIdx.x * 8 + wave;
  if (tile >= (long long)ntiles * mtiles) return;
  int mt = (int)(tile / ntiles);
  int nt = (int)(tile % ntiles);
  const bf16* Ab = A + (long long)blockIdx.y * aBatch;
  const bf16* Bb = B + (long long)blockIdx.y * bBatch;
  float*      Cb = C + (long long)blockIdx.y * cBatch;

  int mrow = mt * 16 + (lane & 15);
  if (mrow >= M) mrow = M - 1;
  long long aoff;
  if (aMode == 1) {
    int img = mrow / rowsPerImg, tt = mrow % rowsPerImg;
    aoff = ((long long)img * padT + tt) * rowStride;
  } else {
    aoff = (long long)mrow * K;
  }
  int ncol = nt * 16 + (lane & 15);
  int ncl  = ncol < N ? ncol : N - 1;
  int half = lane >> 4;
  int abase = half * 8;
  int kb    = half * 16;

  f32x8 acc;
  #pragma unroll
  for (int e = 0; e < 8; ++e) acc[e] = 0.0f;

  int ksteps = (K + 31) >> 5;
  for (int kt = 0; kt < ksteps; ++kt) {
    int k0 = kt * 32;
    bf16x16 a, b;
    #pragma unroll
    for (int e = 0; e < 8; ++e) {
      int ka  = k0 + kb + e;
      int ka2 = ka + 8;
      a[e]     = (ka  < K) ? Ab[aoff + ka ] : (bf16)0.0f;
      a[e + 8] = (ka2 < K) ? Ab[aoff + ka2] : (bf16)0.0f;
    }
    #pragma unroll
    for (int e = 0; e < 16; ++e) {
      int kk = k0 + kb + e;
      b[e] = (kk < K) ? Bb[(long long)ncl * K + kk] : (bf16)0.0f;
    }
    acc = __builtin_amdgcn_wmma_f32_16x16x32_bf16(
        false, a, false, b, (short)0, acc, false, false);
  }
  float bv = bias ? bias[ncl] : 0.0f;
  #pragma unroll
  for (int r = 0; r < 8; ++r) {
    int mm = mt * 16 + r + half * 8;
    if (mm < M && ncol < N)
      Cb[(long long)mm * N + ncol] = acc[r] + bv;
  }
}

// ---------------- residuals / transposes -----------------------------------

// out[b,c,t,f] = Cct[row(b,t|f)*48 + c] + prev[b,c,t,f]
__global__ __launch_bounds__(256) void k_add_res(
    const float* __restrict__ Cct, const float* __restrict__ prev,
    float* __restrict__ out, int mode)
{
  long long idx = (long long)blockIdx.x * 256 + threadIdx.x;
  if (idx >= (long long)Bd * Cd * Td * Fd) return;
  int f = (int)(idx % Fd);
  int t = (int)((idx / Fd) % Td);
  int c = (int)((idx / ((long long)Fd * Td)) % Cd);
  int b = (int)(idx / ((long long)Fd * Td * Cd));
  long long r = (mode == 0) ? ((long long)(b * Td + t) * Fd + f)
                            : ((long long)(b * Fd + f) * Td + t);
  out[idx] = Cct[r * Cd + c] + prev[idx];
}

// interT[((b,t,f))*48 + c] = bf16(inter[b,c,t,f])
__global__ __launch_bounds__(256) void k_interT(
    const float* __restrict__ inter, bf16* __restrict__ o)
{
  long long idx = (long long)blockIdx.x * 256 + threadIdx.x;
  if (idx >= (long long)MR * Cd) return;
  int c = (int)(idx % Cd);
  long long r = idx / Cd;
  int f = (int)(r % Fd), t = (int)((r / Fd) % Td), b = (int)(r / ((long long)Fd * Td));
  o[idx] = (bf16)inter[(((size_t)b * Cd + c) * Td + t) * Fd + f];
}

// A2[((b,t,f))*48 + h*12+o] = bf16(Oatt[(h,b)][t][o*65+f])
__global__ __launch_bounds__(256) void k_build_A2(
    const float* __restrict__ Oatt, bf16* __restrict__ o)
{
  long long idx = (long long)blockIdx.x * 256 + threadIdx.x;
  if (idx >= (long long)MR * Cd) return;
  int cch = (int)(idx % Cd);
  long long r = idx / Cd;
  int h = cch / CVd, oo = cch % CVd;
  int f = (int)(r % Fd), t = (int)((r / Fd) % Td), b = (int)(r / ((long long)Fd * Td));
  o[idx] = (bf16)Oatt[(((size_t)(h * Bd + b)) * Td + t) * DV + oo * Fd + f];
}

// ---------------- head projection post (PReLU + LN over (o,f)) -------------
__global__ __launch_bounds__(256) void k_head_post(
    const float* __restrict__ Cmat, int Ncols, int Eh,
    const float* __restrict__ alpha, const float* __restrict__ gg,
    const float* __restrict__ gb, bf16* __restrict__ outp, int transposed)
{
  __shared__ float sh[256];
  int id = blockIdx.x;
  int t = id % Td, b = (id / Td) % Bd, h = id / (Td * Bd);
  float a = alpha[h];
  int nElem = Eh * Fd;
  float s1 = 0.f, s2 = 0.f;
  for (int i = threadIdx.x; i < nElem; i += 256) {
    int o = i / Fd, f = i % Fd;
    long long r = (long long)(b * Td + t) * Fd + f;
    float v = Cmat[r * Ncols + h * Eh + o];
    v = v >= 0.f ? v : a * v;
    s1 += v; s2 += v * v;
  }
  s1 = blockSum(s1, sh);
  s2 = blockSum(s2, sh);
  float mu = s1 / nElem;
  float var = s2 / nElem - mu * mu;
  float rs = rsqrtf(var + 1e-5f);
  long long batch = (long long)(h * Bd + b);
  for (int i = threadIdx.x; i < nElem; i += 256) {
    int o = i / Fd, f = i % Fd;
    long long r = (long long)(b * Td + t) * Fd + f;
    float v = Cmat[r * Ncols + h * Eh + o];
    v = v >= 0.f ? v : a * v;
    v = (v - mu) * rs * gg[(h * Eh + o) * Fd + f] + gb[(h * Eh + o) * Fd + f];
    if (!transposed)
      outp[(batch * Td + t) * (long long)nElem + (o * Fd + f)] = (bf16)v;
    else
      outp[(batch * (long long)nElem + (o * Fd + f)) * Td + t] = (bf16)v;
  }
}

// ---------------- softmax over 500 keys, write bf16 ------------------------
__global__ __launch_bounds__(256) void k_softmax(
    const float* __restrict__ S, bf16* __restrict__ O)
{
  __shared__ float sh[256];
  long long base = (long long)blockIdx.x * Td;
  const float scale = rsqrtf((float)DQ);   // 1/sqrt(520)
  float m = -1e30f;
  for (int i = threadIdx.x; i < Td; i += 256) m = fmaxf(m, S[base + i] * scale);
  m = blockMax(m, sh);
  float s = 0.f;
  for (int i = threadIdx.x; i < Td; i += 256) s += __expf(S[base + i] * scale - m);
  s = blockSum(s, sh);
  float inv = 1.0f / s;
  for (int i = threadIdx.x; i < Td; i += 256)
    O[base + i] = (bf16)(__expf(S[base + i] * scale - m) * inv);
}

// ---------------- final PReLU + LN over (c,f) + residual -------------------
__global__ __launch_bounds__(256) void k_final_post(
    const float* __restrict__ Cf, const float* __restrict__ apv,
    const float* __restrict__ gg, const float* __restrict__ gb,
    const float* __restrict__ inter, float* __restrict__ out)
{
  __shared__ float sh[256];
  int id = blockIdx.x;
  int t = id % Td, b = id / Td;
  float a = apv[0];
  const int nElem = Cd * Fd;   // 3120
  float s1 = 0.f, s2 = 0.f;
  for (int i = threadIdx.x; i < nElem; i += 256) {
    int c = i / Fd, f = i % Fd;
    long long r = (long long)(b * Td + t) * Fd + f;
    float v = Cf[r * Cd + c];
    v = v >= 0.f ? v : a * v;
    s1 += v; s2 += v * v;
  }
  s1 = blockSum(s1, sh);
  s2 = blockSum(s2, sh);
  float mu = s1 / nElem;
  float var = s2 / nElem - mu * mu;
  float rs = rsqrtf(var + 1e-5f);
  for (int i = threadIdx.x; i < nElem; i += 256) {
    int c = i / Fd, f = i % Fd;
    long long r = (long long)(b * Td + t) * Fd + f;
    float v = Cf[r * Cd + c];
    v = v >= 0.f ? v : a * v;
    long long oidx = (((long long)b * Cd + c) * Td + t) * Fd + f;
    out[oidx] = (v - mu) * rs * gg[c * Fd + f] + gb[c * Fd + f] + inter[oidx];
  }
}

// ---------------------------------------------------------------------------
static inline unsigned cdivu(long long a, long long b){ return (unsigned)((a + b - 1) / b); }

extern "C" void kernel_launch(void* const* d_in, const int* in_sizes, int n_in,
                              void* d_out, int out_size, void* d_ws, size_t ws_size,
                              hipStream_t stream)
{
  (void)in_sizes; (void)n_in; (void)out_size; (void)ws_size;
  const float* x        = (const float*)d_in[0];
  const float* g_intra  = (const float*)d_in[2];
  const float* b_intra  = (const float*)d_in[3];
  const float* Wih_if   = (const float*)d_in[4];
  const float* Whh_if   = (const float*)d_in[5];
  const float* b_if     = (const float*)d_in[6];
  const float* Wih_ib   = (const float*)d_in[7];
  const float* Whh_ib   = (const float*)d_in[8];
  const float* b_ib     = (const float*)d_in[9];
  const float* Wt_intra = (const float*)d_in[10];
  const float* bt_intra = (const float*)d_in[11];
  const float* g_inter  = (const float*)d_in[12];
  const float* b_inter  = (const float*)d_in[13];
  const float* Wih_ef   = (const float*)d_in[14];
  const float* Whh_ef   = (const float*)d_in[15];
  const float* b_ef     = (const float*)d_in[16];
  const float* Wih_eb   = (const float*)d_in[17];
  const float* Whh_eb   = (const float*)d_in[18];
  const float* b_eb     = (const float*)d_in[19];
  const float* Wt_inter = (const float*)d_in[20];
  const float* bt_inter = (const float*)d_in[21];
  const float* Wq  = (const float*)d_in[22]; const float* bq  = (const float*)d_in[23];
  const float* aq  = (const float*)d_in[24]; const float* gqg = (const float*)d_in[25];
  const float* gqb = (const float*)d_in[26];
  const float* Wk  = (const float*)d_in[27]; const float* bk  = (const float*)d_in[28];
  const float* ak  = (const float*)d_in[29]; const float* gkg = (const float*)d_in[30];
  const float* gkb = (const float*)d_in[31];
  const float* Wv  = (const float*)d_in[32]; const float* bv  = (const float*)d_in[33];
  const float* av  = (const float*)d_in[34]; const float* gvg = (const float*)d_in[35];
  const float* gvb = (const float*)d_in[36];
  const float* Wp  = (const float*)d_in[37]; const float* bp  = (const float*)d_in[38];
  const float* ap  = (const float*)d_in[39]; const float* gpg = (const float*)d_in[40];
  const float* gpb = (const float*)d_in[41];
  float* out = (float*)d_out;

  // bump allocator over workspace
  char* wsb = (char*)d_ws;
  size_t cur = 0;
  auto alloc = [&](size_t bytes) -> char* {
    char* p = wsb + cur;
    cur += (bytes + 255) & ~(size_t)255;
    return p;
  };
  const long long ULN  = (long long)Bd * Cd * Td * Fd;      // 3,120,000
  float* uln   = (float*)alloc((size_t)ULN * 4);            // reused for inter-path LN
  bf16*  Uf    = (bf16*) alloc((size_t)NSB * LB * INd * 2); // max of intra/inter
  bf16*  Ub    = (bf16*) alloc((size_t)NSB * LB * INd * 2);
  bf16*  wIF   = (bf16*) alloc((size_t)G4 * 384 * 2);
  bf16*  wIB   = (bf16*) alloc((size_t)G4 * 384 * 2);
  bf16*  wEF   = (bf16*) alloc((size_t)G4 * 384 * 2);
  bf16*  wEB   = (bf16*) alloc((size_t)G4 * 384 * 2);
  bf16*  WcatA = (bf16*) alloc((size_t)Cd * 1536 * 2);
  bf16*  WcatB = (bf16*) alloc((size_t)Cd * 1536 * 2);
  bf16*  WqB   = (bf16*) alloc((size_t)NHd * Ed * Cd * 2);
  bf16*  WkB   = (bf16*) alloc((size_t)NHd * Ed * Cd * 2);
  bf16*  WvB   = (bf16*) alloc((size_t)NHd * CVd * Cd * 2);
  bf16*  WpB   = (bf16*) alloc((size_t)Cd * Cd * 2);
  const long long XPAD = (long long)NSA * PTA * 384;        // >= NSB*PTB*384
  bf16*  Xpad  = (bf16*) alloc((size_t)XPAD * 2);
  float* Cct   = (float*)alloc((size_t)MR * Cd * 4);        // reused: convT / Cv / Cfin
  float* intra = (float*)alloc((size_t)ULN * 4);
  float* inter = (float*)alloc((size_t)ULN * 4);
  bf16*  interT= (bf16*) alloc((size_t)MR * Cd * 2);
  float* Cq    = (float*)alloc((size_t)MR * 32 * 4);
  float* Ck    = (float*)alloc((size_t)MR * 32 * 4);
  bf16*  qf    = (bf16*) alloc((size_t)8 * Td * DQ * 2);
  bf16*  kf    = (bf16*) alloc((size_t)8 * Td * DQ * 2);
  bf16*  vfT   = (bf16*) alloc((size_t)8 * DV * Td * 2);
  float* scores= (float*)alloc((size_t)8 * Td * Td * 4);
  bf16*  attnB = (bf16*) alloc((size_t)8 * Td * Td * 2);
  float* Oatt  = (float*)alloc((size_t)8 * Td * DV * 4);
  bf16*  A2    = (bf16*) alloc((size_t)MR * Cd * 2);

  auto gemmGrid = [](int M, int N, int batch) {
    long long tiles = (long long)((M + 15) / 16) * ((N + 15) / 16);
    return dim3(cdivu(tiles, 8), (unsigned)batch, 1);
  };

  // ---- weight preparation -------------------------------------------------
  unsigned gw = cdivu((long long)G4 * 384, 256);
  k_combine_w<<<gw, 256, 0, stream>>>(Wih_if, Whh_if, wIF);
  k_combine_w<<<gw, 256, 0, stream>>>(Wih_ib, Whh_ib, wIB);
  k_combine_w<<<gw, 256, 0, stream>>>(Wih_ef, Whh_ef, wEF);
  k_combine_w<<<gw, 256, 0, stream>>>(Wih_eb, Whh_eb, wEB);
  unsigned gc = cdivu((long long)Cd * 1536, 256);
  k_wcat<<<gc, 256, 0, stream>>>(Wt_intra, WcatA);
  k_wcat<<<gc, 256, 0, stream>>>(Wt_inter, WcatB);
  k_f2b<<<cdivu(NHd * Ed * Cd, 256), 256, 0, stream>>>(Wq, WqB, NHd * Ed * Cd);
  k_f2b<<<cdivu(NHd * Ed * Cd, 256), 256, 0, stream>>>(Wk, WkB, NHd * Ed * Cd);
  k_f2b<<<cdivu(NHd * CVd * Cd, 256), 256, 0, stream>>>(Wv, WvB, NHd * CVd * Cd);
  k_f2b<<<cdivu(Cd * Cd, 256), 256, 0, stream>>>(Wp, WpB, Cd * Cd);

  // ---- intra path ---------------------------------------------------------
  k_ln_channel<<<cdivu(MR, 256), 256, 0, stream>>>(x, g_intra, b_intra, uln);
  long long nUA = (long long)NSA * LA * INd;
  k_unfold_intra<<<cdivu(nUA, 256), 256, 0, stream>>>(uln, Uf, 0);
  k_unfold_intra<<<cdivu(nUA, 256), 256, 0, stream>>>(uln, Ub, 1);
  k_zero_bf16<<<cdivu(XPAD, 256), 256, 0, stream>>>(Xpad, XPAD);
  k_lstm_scan<<<cdivu(NSA, 16), 192, 0, stream>>>(Uf, wIF, b_if, Xpad, NSA, LA, PTA, 0, 0);
  k_lstm_scan<<<cdivu(NSA, 16), 192, 0, stream>>>(Ub, wIB, b_ib, Xpad, NSA, LA, PTA, 1, HLd);
  k_gemm<<<gemmGrid(MR, Cd, 1), 256, 0, stream>>>(
      MR, Cd, 1536, Xpad, 0, 1, Fd, PTA, 384, WcatA, 0, Cct, 0, bt_intra);
  k_add_res<<<cdivu(ULN, 256), 256, 0, stream>>>(Cct, x, intra, 0);

  // ---- inter path ---------------------------------------------------------
  k_ln_channel<<<cdivu(MR, 256), 256, 0, stream>>>(intra, g_inter, b_inter, uln);
  long long nUB = (long long)NSB * LB * INd;
  k_unfold_inter<<<cdivu(nUB, 256), 256, 0, stream>>>(uln, Uf, 0);
  k_unfold_inter<<<cdivu(nUB, 256), 256, 0, stream>>>(uln, Ub, 1);
  k_zero_bf16<<<cdivu(XPAD, 256), 256, 0, stream>>>(Xpad, XPAD);
  k_lstm_scan<<<cdivu(NSB, 16), 192, 0, stream>>>(Uf, wEF, b_ef, Xpad, NSB, LB, PTB, 0, 0);
  k_lstm_scan<<<cdivu(NSB, 16), 192, 0, stream>>>(Ub, wEB, b_eb, Xpad, NSB, LB, PTB, 1, HLd);
  k_gemm<<<gemmGrid(MR, Cd, 1), 256, 0, stream>>>(
      MR, Cd, 1536, Xpad, 0, 1, Td, PTB, 384, WcatB, 0, Cct, 0, bt_inter);
  k_add_res<<<cdivu(ULN, 256), 256, 0, stream>>>(Cct, intra, inter, 1);

  // ---- attention ----------------------------------------------------------
  k_interT<<<cdivu((long long)MR * Cd, 256), 256, 0, stream>>>(inter, interT);
  k_gemm<<<gemmGrid(MR, 32, 1), 256, 0, stream>>>(
      MR, 32, Cd, interT, 0, 0, 0, 0, 0, WqB, 0, Cq, 0, bq);
  k_gemm<<<gemmGrid(MR, 32, 1), 256, 0, stream>>>(
      MR, 32, Cd, interT, 0, 0, 0, 0, 0, WkB, 0, Ck, 0, bk);
  k_gemm<<<gemmGrid(MR, Cd, 1), 256, 0, stream>>>(
      MR, Cd, Cd, interT, 0, 0, 0, 0, 0, WvB, 0, Cct, 0, bv);
  k_head_post<<<NHd * Bd * Td, 256, 0, stream>>>(Cq, 32, Ed, aq, gqg, gqb, qf, 0);
  k_head_post<<<NHd * Bd * Td, 256, 0, stream>>>(Ck, 32, Ed, ak, gkg, gkb, kf, 0);
  k_head_post<<<NHd * Bd * Td, 256, 0, stream>>>(Cct, Cd, CVd, av, gvg, gvb, vfT, 1);
  k_gemm<<<gemmGrid(Td, Td, 8), 256, 0, stream>>>(
      Td, Td, DQ, qf, (long long)Td * DQ, 0, 0, 0, 0,
      kf, (long long)Td * DQ, scores, (long long)Td * Td, (const float*)nullptr);
  k_softmax<<<8 * Td, 256, 0, stream>>>(scores, attnB);
  k_gemm<<<gemmGrid(Td, DV, 8), 256, 0, stream>>>(
      Td, DV, Td, attnB, (long long)Td * Td, 0, 0, 0, 0,
      vfT, (long long)DV * Td, Oatt, (long long)Td * DV, (const float*)nullptr);
  k_build_A2<<<cdivu((long long)MR * Cd, 256), 256, 0, stream>>>(Oatt, A2);
  k_gemm<<<gemmGrid(MR, Cd, 1), 256, 0, stream>>>(
      MR, Cd, Cd, A2, 0, 0, 0, 0, 0, WpB, 0, Cct, 0, bp);   // Cct reused as Cfin
  k_final_post<<<Bd * Td, 256, 0, stream>>>(Cct, ap, gpg, gpb, inter, out);
}
